// GAT_16277926052539
// MI455X (gfx1250) — matrix operand order repaired and checked by reading, not compiled, not run on hardware
//
#include <hip/hip_runtime.h>
#include <hip/hip_bf16.h>

typedef __attribute__((ext_vector_type(16))) __bf16 v16bf;
typedef __attribute__((ext_vector_type(8)))  float  v8f;

#define GAT_ALPHA 0.2f
#define CAP 1024   // max neighbors per row (degree ~ Binom(8192, 32/8192); max << 1024)

// ---------------------------------------------------------------------------
// Wave-tile bf16 WMMA GEMM: C[head][M][ldc] = A[M][lda] @ B[head][K][ldb]
// One wave (32 lanes) produces one 16x16 f32 tile, looping K in steps of 32.
// VGPR layouts per CDNA5 ISA 7.12.2 (16-bit A 16x32, 16-bit B 32x16, f32 C 16x16).
// ---------------------------------------------------------------------------
__global__ __launch_bounds__(256)
void wmma_gemm_bf16(const float* __restrict__ A, int lda,
                    const float* __restrict__ B0, long long bHeadStride, int ldb,
                    float* __restrict__ C0, long long cHeadStride, int ldc,
                    int M, int Hper, int heads, int Kdim)
{
    int wave = (int)((blockIdx.x * blockDim.x + threadIdx.x) >> 5);
    int lane = (int)(threadIdx.x & 31);
    int ntH   = Hper >> 4;            // n-tiles per head
    int tilesN = ntH * heads;
    int tilesM = M >> 4;
    if (wave >= tilesM * tilesN) return;
    int tm   = wave / tilesN;
    int tn   = wave % tilesN;
    int head = tn / ntH;
    int tnh  = tn % ntH;

    const float* B = B0 + (long long)head * bHeadStride;
    float*       C = C0 + (long long)head * cHeadStride;

    int g  = lane >> 4;               // half-wave group
    int lm = lane & 15;
    int rowA = tm * 16 + lm;          // A row for this lane
    int colB = tnh * 16 + lm;         // B/C column for this lane

    v8f acc = {0.f, 0.f, 0.f, 0.f, 0.f, 0.f, 0.f, 0.f};

    for (int kb = 0; kb < Kdim; kb += 32) {
        // A: lane needs K = kb + (e/8)*16 + g*8 + (e%8) -> two 8-float runs
        const float* ap = A + (size_t)rowA * lda + kb + g * 8;
        // B: lane needs K = kb + (e/8)*16 + g*8 + (e%8) at column colB
        const float* bp = B + (size_t)(kb + g * 8) * ldb + colB;

        v16bf a, b;
#pragma unroll
        for (int e = 0; e < 8; ++e) {
            a[e]     = (__bf16)ap[e];
            a[8 + e] = (__bf16)ap[16 + e];
        }
#pragma unroll
        for (int e = 0; e < 16; ++e)
            b[e] = (__bf16)bp[(size_t)((e & 7) + (e >> 3) * 16) * ldb];

        acc = __builtin_amdgcn_wmma_f32_16x16x32_bf16(
                false, a, false, b, (short)0, acc, false, false);
    }

    // C/D: lane holds column colB, rows tm*16 + g*8 + r
#pragma unroll
    for (int r = 0; r < 8; ++r)
        C[(size_t)(tm * 16 + g * 8 + r) * ldc + colB] = acc[r];
}

// ---------------------------------------------------------------------------
// f_src/f_dst projections: one wave32 per (head, node); shuffle tree reduce.
// ---------------------------------------------------------------------------
__global__ __launch_bounds__(256)
void fproj(const float* __restrict__ h, const float* __restrict__ asrc,
           const float* __restrict__ adst, float* __restrict__ fs,
           float* __restrict__ fd, int N, int D, int heads)
{
    int wave = (int)((blockIdx.x * blockDim.x + threadIdx.x) >> 5);
    int lane = (int)(threadIdx.x & 31);
    if (wave >= N * heads) return;
    int k = wave / N;
    int n = wave % N;
    const float* hp = h + ((size_t)k * N + n) * D;
    float ss = 0.f, sd = 0.f;
    for (int d = lane; d < D; d += 32) {
        float v = hp[d];
        ss += v * asrc[k * D + d];
        sd += v * adst[k * D + d];
    }
    for (int off = 16; off > 0; off >>= 1) {
        ss += __shfl_xor(ss, off, 32);
        sd += __shfl_xor(sd, off, 32);
    }
    if (lane == 0) {
        fs[(size_t)k * N + n] = ss;
        fd[(size_t)k * N + n] = sd;
    }
}

// ---------------------------------------------------------------------------
// Sparse masked softmax + aggregation. One 256-thread block per node row.
// Deterministic: neighbor compaction via exclusive scan (ascending j),
// fixed-shape tree reductions for max / sum / replica accumulation.
// hmat: [heads][N][D]. relu_out: layer1 writes relu(val) to [N][heads*D];
// layer2 writes val to [N][D].
// ---------------------------------------------------------------------------
__global__ __launch_bounds__(256)
void attn_sparse(const unsigned char* __restrict__ adj, int N,
                 const float* __restrict__ hmat,
                 const float* __restrict__ fs, const float* __restrict__ fd,
                 float* __restrict__ outp, int heads, int D, int relu_out)
{
    __shared__ unsigned short nbr[CAP];
    __shared__ float wts[2 * CAP];
    __shared__ float red[256];
    __shared__ int   counts[256];
    __shared__ float sdenom[2];

    int i = (int)blockIdx.x;
    int t = (int)threadIdx.x;
    const unsigned char* row = adj + (size_t)i * N;

    // ---- scan 8192 bool bytes: 32 contiguous bytes per thread ----
    int base = t * 32;
    __align__(16) unsigned char loc[32];
    *(uint4*)(loc)      = *(const uint4*)(row + base);
    *(uint4*)(loc + 16) = *(const uint4*)(row + base + 16);
    int c = 0;
#pragma unroll
    for (int j = 0; j < 32; ++j) c += (loc[j] != 0);

    counts[t] = c;
    __syncthreads();
    for (int s = 1; s < 256; s <<= 1) {          // inclusive Hillis-Steele scan
        int tmp = (t >= s) ? counts[t - s] : 0;
        __syncthreads();
        counts[t] += tmp;
        __syncthreads();
    }
    int off = counts[t] - c;                      // exclusive offset
    int cnt = counts[255];
    if (cnt > CAP) cnt = CAP;
#pragma unroll
    for (int j = 0; j < 32; ++j) {
        if (loc[j]) {
            if (off < CAP) nbr[off] = (unsigned short)(base + j);
            off++;
        }
    }
    __syncthreads();

    // ---- per-head masked softmax over neighbor list ----
    for (int k = 0; k < heads; ++k) {
        float fsv = fs[(size_t)k * N + i];
        float lmax = -3.0e38f;
        for (int idx = t; idx < cnt; idx += 256) {
            float z = fsv + fd[(size_t)k * N + nbr[idx]];
            float e = (z >= 0.f) ? z : GAT_ALPHA * z;
            lmax = fmaxf(lmax, e);
        }
        red[t] = lmax; __syncthreads();
        for (int s = 128; s > 0; s >>= 1) {
            if (t < s) red[t] = fmaxf(red[t], red[t + s]);
            __syncthreads();
        }
        float m = red[0];
        __syncthreads();

        float lsum = 0.f;
        for (int idx = t; idx < cnt; idx += 256) {
            float z = fsv + fd[(size_t)k * N + nbr[idx]];
            float e = (z >= 0.f) ? z : GAT_ALPHA * z;
            float w = __expf(e - m);
            wts[k * CAP + idx] = w;
            lsum += w;
        }
        red[t] = lsum; __syncthreads();
        for (int s = 128; s > 0; s >>= 1) {
            if (t < s) red[t] += red[t + s];
            __syncthreads();
        }
        if (t == 0) sdenom[k] = red[0];
        __syncthreads();
    }

    // ---- aggregation: thread -> (replica r, head kk, dim d) ----
    int d  = t % D;
    int kk = (t / D) % heads;
    int DH = D * heads;
    int r  = t / DH;
    int R  = 256 / DH;
    float acc = 0.f;
    for (int idx = r; idx < cnt; idx += R) {
        int j = nbr[idx];
        acc += wts[kk * CAP + idx] * hmat[((size_t)kk * N + j) * D + d];
    }
    if (R > 1) {
        red[t] = acc; __syncthreads();
        for (int s = R >> 1; s > 0; s >>= 1) {
            if (r < s) red[t] += red[t + s * DH];
            __syncthreads();
        }
        acc = red[t];
    }
    if (r == 0) {
        float val = acc / sdenom[kk];
        if (relu_out)  // relu(elu(x)) == relu(x)
            outp[(size_t)i * DH + kk * D + d] = fmaxf(val, 0.f);
        else           // mean over 1 head == identity
            outp[(size_t)i * D + d] = val;
    }
}

// ---------------------------------------------------------------------------
extern "C" void kernel_launch(void* const* d_in, const int* in_sizes, int n_in,
                              void* d_out, int out_size, void* d_ws, size_t ws_size,
                              hipStream_t stream)
{
    (void)in_sizes; (void)n_in; (void)out_size; (void)ws_size;

    const float*         x    = (const float*)d_in[0];
    const unsigned char* adj  = (const unsigned char*)d_in[1];  // jax bool = 1 byte
    const float*         W1   = (const float*)d_in[2];
    const float*         a1s  = (const float*)d_in[3];
    const float*         a1d  = (const float*)d_in[4];
    const float*         W2   = (const float*)d_in[5];
    const float*         a2s  = (const float*)d_in[6];
    const float*         a2d  = (const float*)d_in[7];
    float*               out  = (float*)d_out;

    const int N = 8192, F = 512, H = 128, K = 2, C = 16, KH = 256;

    float* ws  = (float*)d_ws;
    float* h1  = ws;                          // [2][8192][128]
    float* f1s = h1 + (size_t)K * N * H;      // [2][8192]
    float* f1d = f1s + (size_t)K * N;
    float* h1r = f1d + (size_t)K * N;         // [8192][256]  (layer-1 relu output)
    float* h2  = h1r + (size_t)N * KH;        // [8192][16]
    float* f2s = h2 + (size_t)N * C;
    float* f2d = f2s + N;

    // Layer 1 GEMM: 512 m-tiles * 16 n-tiles = 8192 waves
    {
        int waves = (N / 16) * (H / 16) * K;
        wmma_gemm_bf16<<<(waves * 32 + 255) / 256, 256, 0, stream>>>(
            x, F, W1, (long long)F * H, H, h1, (long long)N * H, H, N, H, K, F);
    }
    {
        int waves = K * N;
        fproj<<<(waves * 32 + 255) / 256, 256, 0, stream>>>(h1, a1s, a1d, f1s, f1d, N, H, K);
    }
    attn_sparse<<<N, 256, 0, stream>>>(adj, N, h1, f1s, f1d, h1r, K, H, 1);

    // Layer 2 GEMM: 512 m-tiles * 1 n-tile
    {
        int waves = (N / 16) * (C / 16);
        wmma_gemm_bf16<<<(waves * 32 + 255) / 256, 256, 0, stream>>>(
            h1r, KH, W2, 0LL, C, h2, 0LL, C, N, C, 1, KH);
    }
    {
        int waves = N;
        fproj<<<(waves * 32 + 255) / 256, 256, 0, stream>>>(h2, a2s, a2d, f2s, f2d, N, C, 1);
    }
    attn_sparse<<<N, 256, 0, stream>>>(adj, N, h2, f2s, f2d, out, 1, C, 0);
}
